// GCU_56401510531237
// MI455X (gfx1250) — matrix operand order repaired and checked by reading, not compiled, not run on hardware
//
#include <hip/hip_runtime.h>

// Problem constants (B,C,H,W = 4,128,64,64 ; K = 32)
#define B_ 4
#define C_ 128
#define N_ 4096
#define K_ 32

typedef __attribute__((ext_vector_type(16))) _Float16 v16h;
typedef __attribute__((ext_vector_type(8)))  _Float16 v8h;
typedef __attribute__((ext_vector_type(8)))  float    v8f;

__device__ __forceinline__ v8f wmma_f16(v16h a, v16h b, v8f c) {
  // D = A(16x32 f16) x B(32x16 f16) + C(16x16 f32)
  return __builtin_amdgcn_wmma_f32_16x16x32_f16(false, a, false, b, (short)0, c,
                                                false, false);
}

// ---------------------------------------------------------------------------
// Kernel 1: precompute.
//  wscale[k*C+c] = 1/(sigmoid(sig)+eps)^2  (f32, for kernel 4)
//  tvec[k]       = sum_c a^2 * w
//  BwSw          = packed B-matrix [w ; -2*a*w] pre-swizzled into WMMA
//                  B-fragment order: [kb(8)][kt(2)][lane(32)][e(16)] halves,
//                  kk = kb*32 + e + 16*(lane>=16), k = (lane&15) + 16*kt.
//  Zeroes atomic accumulators (denom, M1) each call (determinism).
// ---------------------------------------------------------------------------
__global__ __launch_bounds__(256) void prep_kernel(
    const float* __restrict__ anchor, const float* __restrict__ sigma_param,
    float* __restrict__ wscale, _Float16* __restrict__ BwSw,
    float* __restrict__ tvec, float* __restrict__ denom,
    float* __restrict__ M1) {
  __shared__ float tsh[K_];
  const int t = threadIdx.x;
  if (t < K_) tsh[t] = 0.f;
  __syncthreads();
  for (int idx = t; idx < K_ * C_; idx += 256) {
    const int k = idx >> 7, c = idx & 127;
    const float sp = 1.f / (1.f + __expf(-sigma_param[idx])) + 1e-7f;
    const float w  = 1.f / (sp * sp);
    wscale[idx] = w;
    atomicAdd(&tsh[k], anchor[idx] * anchor[idx] * w);
    (void)c;
  }
  // B-fragments in lane-contiguous order (one 32B load per lane in kernel 2)
  for (int idx = t; idx < 2 * C_ * K_; idx += 256) {
    const int e    = idx & 15;
    const int lane = (idx >> 4) & 31;
    const int kt   = (idx >> 9) & 1;
    const int kb   = idx >> 10;                       // 0..7
    const int kk   = kb * 32 + e + 16 * (lane >> 4);  // reduction index 0..255
    const int k    = (lane & 15) + 16 * kt;
    const int c    = (kk < C_) ? kk : (kk - C_);
    const float sp = 1.f / (1.f + __expf(-sigma_param[k * C_ + c])) + 1e-7f;
    const float w  = 1.f / (sp * sp);
    BwSw[idx] = (_Float16)((kk < C_) ? w : (-2.f * anchor[k * C_ + c] * w));
  }
  __syncthreads();
  if (t < K_) tvec[t] = tsh[t];
  for (int idx = t; idx < B_ * K_; idx += 256) denom[idx] = 0.f;
  for (int idx = t; idx < B_ * K_ * C_; idx += 256) M1[idx] = 0.f;
}

// ---------------------------------------------------------------------------
// Kernel 2: logits = -0.5*(X2@w - 2X@u + t) via WMMA, + softmax over K.
// grid (N/128, B), 256 thr = 8 waves; each wave owns a 16-row n tile.
// X^2 and X chunks share the same channels -> x loaded once, 4 WMMAs/iter.
// Writes saH (k-major), saT (n-major) and per-(b,k) denominator (atomic).
// ---------------------------------------------------------------------------
__global__ __launch_bounds__(256) void logits_softmax_kernel(
    const float* __restrict__ x, const _Float16* __restrict__ BwSw,
    const float* __restrict__ tvec, _Float16* __restrict__ saH,
    _Float16* __restrict__ saT, float* __restrict__ denom) {
  __shared__ float tsh[K_];
  __shared__ float dnsh[K_];
  __shared__ float ls[8][16 * 33];        // per-wave logit staging (padded)
  const int t = threadIdx.x;
  const int lane = t & 31, wv = t >> 5;
  const int hi = lane >> 4, lm = lane & 15;
  const int b = blockIdx.y;
  if (t < K_) { tsh[t] = tvec[t]; dnsh[t] = 0.f; }
  __syncthreads();

  const int n0 = blockIdx.x * 128 + wv * 16;
  const int nrow = n0 + lm;               // A-fragment row (ISA 7.12.2)
  v8f acc0 = {0.f,0.f,0.f,0.f,0.f,0.f,0.f,0.f};
  v8f acc1 = {0.f,0.f,0.f,0.f,0.f,0.f,0.f,0.f};

  for (int kb2 = 0; kb2 < 4; ++kb2) {     // channels 32 at a time
    v16h aSq, aLin;
#pragma unroll
    for (int e = 0; e < 16; ++e) {
      const int c = kb2 * 32 + e + 16 * hi;   // A K-index map
      const float v = x[(b * C_ + c) * N_ + nrow];
      aSq[e]  = (_Float16)(v * v);
      aLin[e] = (_Float16)v;
    }
    const v16h b0s = *(const v16h*)(BwSw + (((kb2    ) * 2 + 0) * 32 + lane) * 16);
    const v16h b1s = *(const v16h*)(BwSw + (((kb2    ) * 2 + 1) * 32 + lane) * 16);
    const v16h b0l = *(const v16h*)(BwSw + (((kb2 + 4) * 2 + 0) * 32 + lane) * 16);
    const v16h b1l = *(const v16h*)(BwSw + (((kb2 + 4) * 2 + 1) * 32 + lane) * 16);
    acc0 = wmma_f16(aSq,  b0s, acc0);
    acc1 = wmma_f16(aSq,  b1s, acc1);
    acc0 = wmma_f16(aLin, b0l, acc0);
    acc1 = wmma_f16(aLin, b1l, acc1);
  }
  // logits = -0.5*(acc + t[k]); stage to LDS for per-row softmax over K
#pragma unroll
  for (int v = 0; v < 8; ++v) {
    const int m = v + 8 * hi;
    ls[wv][m * 33 + lm]      = -0.5f * (acc0[v] + tsh[lm]);
    ls[wv][m * 33 + lm + 16] = -0.5f * (acc1[v] + tsh[lm + 16]);
  }
  __syncthreads();
  if (lane < 16) {
    const int n = n0 + lane;
    float mx = -1e30f;
#pragma unroll
    for (int k = 0; k < K_; ++k) mx = fmaxf(mx, ls[wv][lane * 33 + k]);
    float ev[K_]; float s = 0.f;
#pragma unroll
    for (int k = 0; k < K_; ++k) {
      ev[k] = __expf(ls[wv][lane * 33 + k] - mx); s += ev[k];
    }
    const float inv = 1.f / s;
    v16h s0, s1;
#pragma unroll
    for (int k = 0; k < K_; ++k) {
      const float sv = ev[k] * inv;
      saH[((b * K_ + k) * N_) + n] = (_Float16)sv;
      if (k < 16) s0[k] = (_Float16)sv; else s1[k - 16] = (_Float16)sv;
      atomicAdd(&dnsh[k], sv);
    }
    _Float16* tp = saT + (size_t)(b * N_ + n) * K_;   // 64B contiguous/thread
    *(v16h*)tp = s0;
    *(v16h*)(tp + 16) = s1;
  }
  __syncthreads();
  if (t < K_) atomicAdd(&denom[b * K_ + t], dnsh[t]);
}

// ---------------------------------------------------------------------------
// Kernel 3: M1[b,k,c] = sum_n sa[b,k,n]*x[b,c,n]   ([32x4096]@[4096x128]).
// grid (8 n-slices, 2 k-tiles * 8 c-tiles, B), 1 wave/block, f32 atomics out.
// A-fragment = two contiguous v8h loads; B-fragment = four float4 loads.
// ---------------------------------------------------------------------------
__global__ __launch_bounds__(32) void numerator_kernel(
    const float* __restrict__ x, const _Float16* __restrict__ saH,
    float* __restrict__ M1) {
  const int lane = threadIdx.x & 31;
  const int hi = lane >> 4, lm = lane & 15;
  const int b = blockIdx.z;
  const int k0 = (blockIdx.y >> 3) * 16;
  const int c0 = (blockIdx.y & 7) * 16;
  v8f acc = {0.f,0.f,0.f,0.f,0.f,0.f,0.f,0.f};
  for (int nc = 0; nc < 16; ++nc) {       // 512-wide n slice, 32 per step
    const int nb = blockIdx.x * 512 + nc * 32;
    // A: rows k, reduction n.  n = nb + e + 16*hi -> two 16B runs.
    const _Float16* ap = saH + ((b * K_ + k0 + lm) * N_) + nb + 16 * hi;
    const v8h alo = *(const v8h*)ap;
    const v8h ahi = *(const v8h*)(ap + 8);
    const v16h a = __builtin_shufflevector(alo, ahi, 0, 1, 2, 3, 4, 5, 6, 7,
                                           8, 9, 10, 11, 12, 13, 14, 15);
    // B: cols c, reduction n.  n = nb + e + 16*hi -> 16 contiguous floats.
    const float* xp = x + ((b * C_ + c0 + lm) * N_) + nb + 16 * hi;
    const float4 f0 = ((const float4*)xp)[0];
    const float4 f1 = ((const float4*)xp)[1];
    const float4 f2 = ((const float4*)xp)[2];
    const float4 f3 = ((const float4*)xp)[3];
    v16h bf;
    bf[0]  = (_Float16)f0.x; bf[1]  = (_Float16)f0.y;
    bf[2]  = (_Float16)f0.z; bf[3]  = (_Float16)f0.w;
    bf[4]  = (_Float16)f1.x; bf[5]  = (_Float16)f1.y;
    bf[6]  = (_Float16)f1.z; bf[7]  = (_Float16)f1.w;
    bf[8]  = (_Float16)f2.x; bf[9]  = (_Float16)f2.y;
    bf[10] = (_Float16)f2.z; bf[11] = (_Float16)f2.w;
    bf[12] = (_Float16)f3.x; bf[13] = (_Float16)f3.y;
    bf[14] = (_Float16)f3.z; bf[15] = (_Float16)f3.w;
    acc = wmma_f16(a, bf, acc);
  }
#pragma unroll
  for (int v = 0; v < 8; ++v)
    atomicAdd(&M1[((b * K_ + k0 + v + 8 * hi) * C_) + c0 + lm], acc[v]);
}

// ---------------------------------------------------------------------------
// Kernel 4: tiny graph math, one block per batch, f32 scalar in LDS.
// nodes -> L2 norms -> scrambled reshape -> adj softmax -> GCN -> relu.
// Output written directly in WMMA A-fragment order:
//   graphSw[b][ctile(8)][lane(32)][e(16)], c=(lane&15)+16*ctile... row=lane&15,
//   k = e + 16*(lane>=16)  (inverse map applied per (k,d)).
// ---------------------------------------------------------------------------
__global__ __launch_bounds__(256) void graph_kernel(
    const float* __restrict__ anchor, const float* __restrict__ gcn_weight,
    const float* __restrict__ wscale, const float* __restrict__ denom,
    const float* __restrict__ M1, _Float16* __restrict__ graphSw) {
  __shared__ float nodes[K_][C_];
  __shared__ float graphS[C_][K_ + 1];
  __shared__ float adj[K_][K_ + 1];
  __shared__ float support[K_][C_];
  __shared__ float gsum;
  const int b = blockIdx.x;
  const int t = threadIdx.x;
  for (int idx = t; idx < K_ * C_; idx += 256) {
    const int k = idx >> 7, c = idx & 127;
    const float dk = denom[b * K_ + k];
    const float w1 = sqrtf(wscale[idx]);  // 1/sigma'
    nodes[k][c] = w1 * (M1[(b * K_ + k) * C_ + c] - anchor[idx] * dk)
                  / (dk + 1e-7f);
  }
  if (t == 0) gsum = 0.f;
  __syncthreads();
  if (t < K_) {                            // per-node L2 normalize over C
    float s = 0.f;
    for (int c = 0; c < C_; ++c) s += nodes[t][c] * nodes[t][c];
    const float inv = 1.f / fmaxf(sqrtf(s), 1e-12f);
    for (int c = 0; c < C_; ++c) nodes[t][c] *= inv;
  }
  __syncthreads();
  float ps = 0.f;                          // global L2 over flattened K*C
  for (int idx = t; idx < K_ * C_; idx += 256) {
    const float v = nodes[idx >> 7][idx & 127];
    ps += v * v;
  }
  atomicAdd(&gsum, ps);
  __syncthreads();
  const float ginv = 1.f / fmaxf(sqrtf(gsum), 1e-12f);
  // scrambled reshape [K*C] -> [C][K]: graph[c][k] = flat[c*K+k]
  for (int idx = t; idx < K_ * C_; idx += 256)
    graphS[idx >> 5][idx & 31] = nodes[idx >> 7][idx & 127] * ginv;
  __syncthreads();
  for (int idx = t; idx < K_ * K_; idx += 256) {
    const int k = idx >> 5, l = idx & 31;
    float s = 0.f;
    for (int c = 0; c < C_; ++c) s += graphS[c][k] * graphS[c][l];
    adj[k][l] = s;
  }
  __syncthreads();
  if (t < K_) {                            // softmax over l
    float mx = -1e30f;
    for (int l = 0; l < K_; ++l) mx = fmaxf(mx, adj[t][l]);
    float s = 0.f;
    for (int l = 0; l < K_; ++l) { const float e = __expf(adj[t][l] - mx); adj[t][l] = e; s += e; }
    const float inv = 1.f / s;
    for (int l = 0; l < K_; ++l) adj[t][l] *= inv;
  }
  __syncthreads();
  for (int idx = t; idx < K_ * C_; idx += 256) {
    const int k = idx >> 7, d = idx & 127;
    float s = 0.f;
    for (int c = 0; c < C_; ++c) s += graphS[c][k] * gcn_weight[c * C_ + d];
    support[k][d] = s;
  }
  __syncthreads();
  for (int idx = t; idx < K_ * C_; idx += 256) {
    const int k = idx >> 7, d = idx & 127;
    float s = 0.f;
    for (int l = 0; l < K_; ++l) s += adj[k][l] * support[l][d];
    // inverse A-fragment map: k -> (e, hi);  d -> (ctile, lane&15)
    const int g = k >> 3;
    const int hi2 = g >> 1;
    const int e = k & 15;
    const int lane2 = (d & 15) + 16 * hi2;
    const int ctile = d >> 4;
    graphSw[(((b * 8 + ctile) * 32 + lane2) * 16) + e] =
        (_Float16)fmaxf(s, 0.f);           // relu
  }
}

// ---------------------------------------------------------------------------
// Kernel 5: out[b,c,n] = graph[C x 32] @ sa[32 x N] -- one WMMA per 16x16 tile.
// grid (N/128, C/16, B), 8 waves/block. A and B fragments are each a single
// contiguous 32B vector load (pre-swizzled graphSw / transposed saT).
// ---------------------------------------------------------------------------
__global__ __launch_bounds__(256) void project_kernel(
    const _Float16* __restrict__ graphSw, const _Float16* __restrict__ saT,
    float* __restrict__ out) {
  const int t = threadIdx.x;
  const int lane = t & 31, wv = t >> 5;
  const int hi = lane >> 4, lm = lane & 15;
  const int b = blockIdx.z;
  const int ctile = blockIdx.y;
  const int c0 = ctile * 16;
  const int n0 = blockIdx.x * 128 + wv * 16;
  const v16h a  = *(const v16h*)(graphSw + ((b * 8 + ctile) * 32 + lane) * 16);
  const v16h bf = *(const v16h*)(saT + (size_t)(b * N_ + n0 + lm) * K_ + 16 * hi);
  v8f acc = {0.f,0.f,0.f,0.f,0.f,0.f,0.f,0.f};
  acc = wmma_f16(a, bf, acc);
#pragma unroll
  for (int v = 0; v < 8; ++v)
    out[((b * C_ + c0 + v + 8 * hi) * N_) + n0 + lm] = acc[v];
}

// ---------------------------------------------------------------------------
extern "C" void kernel_launch(void* const* d_in, const int* in_sizes, int n_in,
                              void* d_out, int out_size, void* d_ws, size_t ws_size,
                              hipStream_t stream) {
  (void)in_sizes; (void)n_in; (void)out_size; (void)ws_size;
  const float* x           = (const float*)d_in[0];
  const float* anchor      = (const float*)d_in[1];
  const float* sigma_param = (const float*)d_in[2];
  const float* gcn_weight  = (const float*)d_in[3];
  float* out = (float*)d_out;

  char* ws = (char*)d_ws;                        // ~2.2 MB total
  float*    wscale  = (float*)(ws + 0);          // K*C f32      (16 KB)
  _Float16* BwSw    = (_Float16*)(ws + 16384);   // 2C*K f16 swizzled (16 KB)
  float*    tvec    = (float*)(ws + 32768);      // K f32
  float*    denom   = (float*)(ws + 33024);      // B*K f32
  float*    M1      = (float*)(ws + 33536);      // B*K*C f32    (64 KB)
  _Float16* saH     = (_Float16*)(ws + 99072);   // B*K*N f16    (1 MB)
  _Float16* saT     = (_Float16*)(ws + 1147648); // B*N*K f16    (1 MB)
  _Float16* graphSw = (_Float16*)(ws + 2196224); // B*C*K f16 swizzled (32 KB)

  prep_kernel<<<1, 256, 0, stream>>>(anchor, sigma_param, wscale, BwSw, tvec,
                                     denom, M1);
  logits_softmax_kernel<<<dim3(N_ / 128, B_), 256, 0, stream>>>(x, BwSw, tvec,
                                                                saH, saT, denom);
  numerator_kernel<<<dim3(8, 16, B_), 32, 0, stream>>>(x, saH, M1);
  graph_kernel<<<B_, 256, 0, stream>>>(anchor, gcn_weight, wscale, denom, M1,
                                       graphSw);
  project_kernel<<<dim3(N_ / 128, C_ / 16, B_), 256, 0, stream>>>(graphSw, saT,
                                                                  out);
}
